// AggregatedAttention_rev_70222715290087
// MI455X (gfx1250) — matrix operand order repaired and checked by reading, not compiled, not run on hardware
//
#include <hip/hip_runtime.h>
#include <hip/hip_bf16.h>
#include <math.h>

// ---------------- problem constants ----------------
#define RES_   56
#define NPIX   3136          // 56*56
#define CH     256
#define HEADS  8
#define HD     32
#define BATCH  4
#define MROWS  (BATCH*NPIX)  // 12544 = 98*128
#define PP     49            // pooled tokens (7*7)
#define L1     9
#define L2     25
#define NLOGIT 83            // 9 + 25 + 49

typedef unsigned short bf16_t;
typedef __attribute__((ext_vector_type(16))) __bf16 v16bf;
typedef __attribute__((ext_vector_type(8)))  float  v8f;

struct B32x2 { uint4 a, b; };   // 32 bytes == sizeof(v16bf)

__device__ __forceinline__ bf16_t f2bf(float f) {
  unsigned int u = __float_as_uint(f);
  unsigned int r = u + 0x7FFFu + ((u >> 16) & 1u);   // RNE
  return (bf16_t)(r >> 16);
}

// ---- CDNA5 async global->LDS copy (ASYNCcnt path), per CDNA5 ISA ----
// vdst = 32-bit LDS byte address (generic shared ptr truncates to LDS offset per ISA
// aperture rules), vaddr = 64-bit global address.
__device__ __forceinline__ void async_g2l_b128(unsigned int lds_addr, const void* gptr) {
  unsigned long long ga = (unsigned long long)(uintptr_t)gptr;
  asm volatile("global_load_async_to_lds_b128 %0, %1, off"
               :: "v"(lds_addr), "v"(ga) : "memory");
}
__device__ __forceinline__ void wait_asynccnt0() {
#if __has_builtin(__builtin_amdgcn_s_wait_asynccnt)
  __builtin_amdgcn_s_wait_asynccnt(0);
#else
  asm volatile("s_wait_asynccnt 0" ::: "memory");
#endif
}

// ---------------- elementwise converts ----------------
__global__ __launch_bounds__(256)
void k_f32_to_bf16(const float* __restrict__ in, bf16_t* __restrict__ out, int n) {
  int i = blockIdx.x * 256 + threadIdx.x;
  if (i < n) out[i] = f2bf(in[i]);
}

// W (K x N, row-major) -> Wt bf16 (N x K, row-major)
__global__ __launch_bounds__(256)
void k_transpose_bf16(const float* __restrict__ W, bf16_t* __restrict__ Wt, int K, int N) {
  int i = blockIdx.x * 256 + threadIdx.x;
  if (i >= K * N) return;
  int o = i / K, k = i - o * K;
  Wt[i] = f2bf(W[(size_t)k * N + o]);
}

// ---------------- WMMA bf16 GEMM:  C[M,N] = A[M,K] @ Bt[N,K]^T + bias, act ----------------
// block = 256 threads = 8 wave32; tile 128(M) x 64(N); each wave 32x32 via 2x2 WMMA 16x16x32.
// Pipeline per K-step: cooperative GLOBAL_LOAD_ASYNC_TO_LDS_B128 of the 128x32 A-tile and
// 64x32 B-tile -> s_wait_asynccnt 0 -> barrier -> ds_load_b128 fragments -> 4x WMMA.
__global__ __launch_bounds__(256)
void k_wmma_gemm(const bf16_t* __restrict__ A,
                 const bf16_t* __restrict__ Bt,
                 const float*  __restrict__ bias,
                 float* __restrict__ C,
                 int M, int N, int K, int act /*0=none,1=gelu*/) {
  __shared__ __align__(16) bf16_t smA[128 * 32];   // 8 KB
  __shared__ __align__(16) bf16_t smB[64 * 32];    // 4 KB

  const int t    = threadIdx.x;
  const int lane = t & 31;
  const int wid  = t >> 5;
  const int wm   = wid & 3;            // 4 wave rows
  const int wn   = wid >> 2;           // 2 wave cols
  const int mb   = blockIdx.x * 128;   // block row base
  const int nb   = blockIdx.y * 64;    // block col base
  const int half = lane >> 4;          // K-half select (per 16-bit A/B VGPR layouts)
  const int l16  = lane & 15;

  const unsigned int smA_base = (unsigned int)(uintptr_t)smA;
  const unsigned int smB_base = (unsigned int)(uintptr_t)smB;

  v8f acc[2][2] = {};
  for (int k0 = 0; k0 < K; k0 += 32) {
    // ---- stage tiles into LDS via async copies (3 x b128 per thread) ----
    {
      // A tile: 128 rows x 32 k = 512 chunks of 8 bf16; thread handles chunks t, t+256
#pragma unroll
      for (int r = 0; r < 2; ++r) {
        int c   = t + r * 256;
        int row = c >> 2, kc = (c & 3) * 8;
        async_g2l_b128(smA_base + (unsigned)(row * 64 + kc * 2),
                       A + (size_t)(mb + row) * K + k0 + kc);
      }
      // B tile: 64 rows x 32 k = 256 chunks; thread handles chunk t
      {
        int row = t >> 2, kc = (t & 3) * 8;
        async_g2l_b128(smB_base + (unsigned)(row * 64 + kc * 2),
                       Bt + (size_t)(nb + row) * K + k0 + kc);
      }
    }
    if (k0 + 32 < K)  // hint next A K-tile toward L2 (global_prefetch_b8)
      __builtin_prefetch(A + (size_t)(mb + l16) * K + k0 + 32, 0, 0);
    wait_asynccnt0();   // this wave's async writes have landed in LDS
    __syncthreads();    // all waves' tiles visible

    // ---- fragments from LDS (ds_load_b128 pairs) ----
    v16bf afrag[2], bfrag[2];
#pragma unroll
    for (int tm = 0; tm < 2; ++tm) {
      // A 16-bit layout: lanes 0-15 rows M, K 0..7 / 16..23; lanes 16-31 K 8..15 / 24..31
      const bf16_t* ap = smA + (wm * 32 + tm * 16 + l16) * 32 + half * 8;
      B32x2 v;
      v.a = *(const uint4*)(ap);
      v.b = *(const uint4*)(ap + 16);
      afrag[tm] = __builtin_bit_cast(v16bf, v);
    }
#pragma unroll
    for (int tn = 0; tn < 2; ++tn) {
      // B 16-bit layout: lanes 0-15 cols hold K 0..7 / 16..23; lanes 16-31 K 8..15 / 24..31
      const bf16_t* bp = smB + (wn * 32 + tn * 16 + l16) * 32 + half * 8;
      B32x2 v;
      v.a = *(const uint4*)(bp);
      v.b = *(const uint4*)(bp + 16);
      bfrag[tn] = __builtin_bit_cast(v16bf, v);
    }
#pragma unroll
    for (int tm = 0; tm < 2; ++tm)
#pragma unroll
      for (int tn = 0; tn < 2; ++tn)
        acc[tm][tn] = __builtin_amdgcn_wmma_f32_16x16x32_bf16(
            false, afrag[tm], false, bfrag[tn],
            (short)0, acc[tm][tn], false, false);

    __syncthreads();    // everyone done reading LDS before next tile overwrites it
  }

  // D layout: VGPR r -> row = r + 8*(lane>=16), col = lane&15
#pragma unroll
  for (int tm = 0; tm < 2; ++tm)
#pragma unroll
    for (int tn = 0; tn < 2; ++tn) {
      const int col = nb + wn * 32 + tn * 16 + l16;
      const float bb = bias[col];
#pragma unroll
      for (int r = 0; r < 8; ++r) {
        const int row = mb + wm * 32 + tm * 16 + half * 8 + r;
        float v = acc[tm][tn][r] + bb;
        if (act == 1) v = 0.5f * v * (1.0f + erff(v * 0.70710678118654752f));
        C[(size_t)row * N + col] = v;
      }
    }
}

// ---------------- per-head L2 normalize (in place, first CH channels of each row) ----------------
__global__ __launch_bounds__(256)
void k_normalize_heads(float* __restrict__ base, int rows, int rowStride) {
  int t = blockIdx.x * 256 + threadIdx.x;
  if (t >= rows * HEADS) return;
  int row = t >> 3, h = t & 7;
  float* p = base + (size_t)row * rowStride + h * HD;
  float s = 0.f;
#pragma unroll
  for (int d = 0; d < HD; ++d) s += p[d] * p[d];
  float inv = 1.0f / fmaxf(sqrtf(s), 1e-12f);
#pragma unroll
  for (int d = 0; d < HD; ++d) p[d] *= inv;
}

// ---------------- 8x8 avg pool + LayerNorm: xsr (B*N,CH) -> xp (B*P,CH) ----------------
__global__ __launch_bounds__(256)
void k_pool_ln(const float* __restrict__ xsr, const float* __restrict__ g,
               const float* __restrict__ b, float* __restrict__ xp) {
  const int bp = blockIdx.x;            // 0 .. B*49-1
  const int bb = bp / PP, p = bp % PP;
  const int py = p / 7, px = p % 7;
  const int c = threadIdx.x;            // 256
  float s = 0.f;
  for (int i = 0; i < 8; ++i)
    for (int j = 0; j < 8; ++j) {
      int n = (py * 8 + i) * RES_ + (px * 8 + j);
      s += xsr[((size_t)bb * NPIX + n) * CH + c];
    }
  float val = s * (1.0f / 64.0f);

  __shared__ float red[CH];
  red[c] = val; __syncthreads();
  for (int st = 128; st > 0; st >>= 1) { if (c < st) red[c] += red[c + st]; __syncthreads(); }
  float mu = red[0] * (1.0f / CH);
  __syncthreads();
  float dv = val - mu;
  red[c] = dv * dv; __syncthreads();
  for (int st = 128; st > 0; st >>= 1) { if (c < st) red[c] += red[c + st]; __syncthreads(); }
  float var = red[0] * (1.0f / CH);
  xp[(size_t)bp * CH + c] = dv * rsqrtf(var + 1e-5f) * g[c] + b[c];
}

// ---------------- tiny GEMM: kvp (B*P,512) = xp (B*P,256) @ Wkv (256,512) + bkv ----------------
__global__ __launch_bounds__(256)
void k_small_gemm_kvp(const float* __restrict__ xp, const float* __restrict__ Wkv,
                      const float* __restrict__ bkv, float* __restrict__ kvp) {
  int row = blockIdx.x;                       // 0 .. B*P-1
  int o = blockIdx.y * 256 + threadIdx.x;     // 0 .. 511
  const float* xr = xp + (size_t)row * CH;
  float s = bkv[o];
  for (int k = 0; k < CH; ++k) s += xr[k] * Wkv[(size_t)k * (2 * CH) + o];
  kvp[(size_t)row * (2 * CH) + o] = s;
}

// ---------------- fused local+pooled attention ----------------
// one block per (b,n): 8 heads x 32 lanes.  qn (B*N,256), kv (B*N,512) with kn|v,
// kvp (B*P,512) with kpn|vp.  Emits bf16 attention output for the WMMA projection.
__global__ __launch_bounds__(256)
void k_attention(const float* __restrict__ qn, const float* __restrict__ kv,
                 const float* __restrict__ kvp,
                 const float* __restrict__ tok1, const float* __restrict__ bias1,
                 const float* __restrict__ tok2, const float* __restrict__ bias2,
                 bf16_t* __restrict__ outbf) {
  const int bn = blockIdx.x;
  const int bb = bn / NPIX, n = bn % NPIX;
  const int y = n / RES_, x = n % RES_;
  const int h = threadIdx.x >> 5, lane = threadIdx.x & 31;

  __shared__ float qs[CH];
  __shared__ float sc[HEADS][NLOGIT];
  __shared__ float tokw[HEADS][L1 + L2];

  qs[threadIdx.x] = qn[(size_t)bn * CH + threadIdx.x];
  __syncthreads();
  const float* qh = qs + h * HD;

  // ---- scores + token logits ----
  for (int idx = lane; idx < NLOGIT; idx += 32) {
    float a;
    if (idx < L1 + L2) {
      int li, dy, dx, l; const float* tk; const float* bs;
      if (idx < L1) { li = idx;      dy = li / 3 - 1; dx = li % 3 - 1; l = L1; tk = tok1 + (size_t)h * HD * L1 + li; bs = bias1 + h * L1 + li; }
      else          { li = idx - L1; dy = li / 5 - 2; dx = li % 5 - 2; l = L2; tk = tok2 + (size_t)h * HD * L2 + li; bs = bias2 + h * L2 + li; }
      float tw = *bs;
#pragma unroll
      for (int d = 0; d < HD; ++d) tw += qh[d] * tk[(size_t)d * l];
      tokw[h][idx] = tw;
      int yy = y + dy, xx = x + dx;
      if (yy < 0 || yy >= RES_ || xx < 0 || xx >= RES_) a = -__builtin_inff();
      else {
        const float* kh = kv + ((size_t)bb * NPIX + yy * RES_ + xx) * (2 * CH) + h * HD;
        float s = 0.f;
#pragma unroll
        for (int d = 0; d < HD; ++d) s += qh[d] * kh[d];
        a = s;
      }
    } else {
      const float* kh = kvp + ((size_t)bb * PP + (idx - L1 - L2)) * (2 * CH) + h * HD;
      float s = 0.f;
#pragma unroll
      for (int d = 0; d < HD; ++d) s += qh[d] * kh[d];
      a = s;
    }
    sc[h][idx] = a;
  }
  __syncthreads();

  // ---- softmax over 83 logits per head (wave32 shuffle reduction) ----
  float m = -__builtin_inff();
  for (int idx = lane; idx < NLOGIT; idx += 32) m = fmaxf(m, sc[h][idx]);
  for (int off = 16; off; off >>= 1) m = fmaxf(m, __shfl_xor(m, off, 32));
  float sum = 0.f;
  for (int idx = lane; idx < NLOGIT; idx += 32) { float e = __expf(sc[h][idx] - m); sc[h][idx] = e; sum += e; }
  for (int off = 16; off; off >>= 1) sum += __shfl_xor(sum, off, 32);
  float inv = 1.0f / sum;
  for (int idx = lane; idx < NLOGIT; idx += 32) sc[h][idx] *= inv;
  __syncthreads();

  // ---- aggregate V (lane == d; coalesced 128B head loads) ----
  const int d = lane;
  float acc = 0.f;
#pragma unroll 1
  for (int idx = 0; idx < L1 + L2; ++idx) {
    int li, dy, dx;
    if (idx < L1) { li = idx;      dy = li / 3 - 1; dx = li % 3 - 1; }
    else          { li = idx - L1; dy = li / 5 - 2; dx = li % 5 - 2; }
    int yy = y + dy, xx = x + dx;
    if (yy < 0 || yy >= RES_ || xx < 0 || xx >= RES_) continue;  // padded v == 0
    float w = sc[h][idx] + tokw[h][idx];
    acc += w * kv[((size_t)bb * NPIX + yy * RES_ + xx) * (2 * CH) + CH + h * HD + d];
  }
  for (int p = 0; p < PP; ++p)
    acc += sc[h][L1 + L2 + p] * kvp[((size_t)bb * PP + p) * (2 * CH) + CH + h * HD + d];

  outbf[(size_t)bn * CH + h * HD + d] = f2bf(acc);
}

// ---------------- host orchestration ----------------
extern "C" void kernel_launch(void* const* d_in, const int* in_sizes, int n_in,
                              void* d_out, int out_size, void* d_ws, size_t ws_size,
                              hipStream_t stream) {
  const float* x     = (const float*)d_in[0];
  const float* Wq    = (const float*)d_in[1];
  const float* bq    = (const float*)d_in[2];
  const float* Wkv   = (const float*)d_in[3];
  const float* bkv   = (const float*)d_in[4];
  const float* Wsr   = (const float*)d_in[5];
  const float* bsr   = (const float*)d_in[6];
  const float* ln_g  = (const float*)d_in[7];
  const float* ln_b  = (const float*)d_in[8];
  const float* tok1  = (const float*)d_in[9];
  const float* bias1 = (const float*)d_in[10];
  const float* tok2  = (const float*)d_in[11];
  const float* bias2 = (const float*)d_in[12];
  const float* Wproj = (const float*)d_in[13];
  const float* bproj = (const float*)d_in[14];
  (void)in_sizes; (void)n_in; (void)out_size; (void)ws_size;

  char* ws = (char*)d_ws;
  size_t off = 0;
  auto take = [&](size_t bytes) -> char* {
    char* p = ws + off;
    off += (bytes + 255) & ~(size_t)255;
    return p;
  };
  bf16_t* xbf    = (bf16_t*)take((size_t)MROWS * CH * sizeof(bf16_t));
  bf16_t* wqT    = (bf16_t*)take((size_t)CH * CH * sizeof(bf16_t));
  bf16_t* wkvT   = (bf16_t*)take((size_t)(2 * CH) * CH * sizeof(bf16_t));
  bf16_t* wsrT   = (bf16_t*)take((size_t)CH * CH * sizeof(bf16_t));
  bf16_t* wprojT = (bf16_t*)take((size_t)CH * CH * sizeof(bf16_t));
  float*  qbuf   = (float*)take((size_t)MROWS * CH * sizeof(float));        // q -> qn
  float*  kvbuf  = (float*)take((size_t)MROWS * 2 * CH * sizeof(float));    // kn|v
  float*  xsr    = (float*)take((size_t)MROWS * CH * sizeof(float));
  float*  xp     = (float*)take((size_t)BATCH * PP * CH * sizeof(float));
  float*  kvpbuf = (float*)take((size_t)BATCH * PP * 2 * CH * sizeof(float)); // kpn|vp
  bf16_t* attnbf = (bf16_t*)take((size_t)MROWS * CH * sizeof(bf16_t));
  float*  out    = (float*)d_out;

  const int nx = MROWS * CH;
  k_f32_to_bf16<<<(nx + 255) / 256, 256, 0, stream>>>(x, xbf, nx);
  k_transpose_bf16<<<(CH * CH + 255) / 256, 256, 0, stream>>>(Wq, wqT, CH, CH);
  k_transpose_bf16<<<(CH * 2 * CH + 255) / 256, 256, 0, stream>>>(Wkv, wkvT, CH, 2 * CH);
  k_transpose_bf16<<<(CH * CH + 255) / 256, 256, 0, stream>>>(Wsr, wsrT, CH, CH);
  k_transpose_bf16<<<(CH * CH + 255) / 256, 256, 0, stream>>>(Wproj, wprojT, CH, CH);

  // big WMMA GEMMs: M = 12544 = 98 * 128
  dim3 blk(256);
  k_wmma_gemm<<<dim3(MROWS / 128, CH / 64), blk, 0, stream>>>(xbf, wqT, bq, qbuf, MROWS, CH, CH, 0);
  k_wmma_gemm<<<dim3(MROWS / 128, (2 * CH) / 64), blk, 0, stream>>>(xbf, wkvT, bkv, kvbuf, MROWS, 2 * CH, CH, 0);
  k_wmma_gemm<<<dim3(MROWS / 128, CH / 64), blk, 0, stream>>>(xbf, wsrT, bsr, xsr, MROWS, CH, CH, 1);

  k_normalize_heads<<<(MROWS * HEADS + 255) / 256, 256, 0, stream>>>(qbuf, MROWS, CH);
  k_normalize_heads<<<(MROWS * HEADS + 255) / 256, 256, 0, stream>>>(kvbuf, MROWS, 2 * CH);

  k_pool_ln<<<BATCH * PP, 256, 0, stream>>>(xsr, ln_g, ln_b, xp);
  k_small_gemm_kvp<<<dim3(BATCH * PP, 2), 256, 0, stream>>>(xp, Wkv, bkv, kvpbuf);
  k_normalize_heads<<<(BATCH * PP * HEADS + 255) / 256, 256, 0, stream>>>(kvpbuf, BATCH * PP, 2 * CH);

  k_attention<<<MROWS, 256, 0, stream>>>(qbuf, kvbuf, kvpbuf, tok1, bias1, tok2, bias2, attnbf);

  k_wmma_gemm<<<dim3(MROWS / 128, CH / 64), blk, 0, stream>>>(attnbf, wprojT, bproj, out, MROWS, CH, CH, 0);
}
